// pooling_50448685858965
// MI455X (gfx1250) — matrix operand order repaired and checked by reading, not compiled, not run on hardware
//
#include <hip/hip_runtime.h>
#include <hip/hip_bf16.h>
#include <stdint.h>

// ---------------------------------------------------------------------------
// Problem constants (match the reference)
// ---------------------------------------------------------------------------
#define NN   4096      // nodes
#define DD   256       // channels
#define EE   65536     // edges
#define UB   3.0f      // UPPER_BOUND
#define SADD 0.5f      // SELF_ADD
#define MW   128       // mask words per node row (4096 bits)

typedef _Float16 half_t;
typedef __attribute__((ext_vector_type(16))) _Float16 v16h;
typedef __attribute__((ext_vector_type(8)))  float    v8f;

// Workspace header (uint32 slots):
// H[0]=min bits  H[1]=max bits  H[2]=U  H[3]=K  H[4]=Enew
// H[5]=o1(x_new) H[6]=o2(S)     H[7]=o3(Ms)  H[8]=o4(edge_index copy)

// ---------------------------------------------------------------------------
// 0) init
// ---------------------------------------------------------------------------
__global__ void init_kernel(uint32_t* H, float* tot, float* cnt) {
    int i = blockIdx.x * 256 + threadIdx.x;
    if (i < NN) { tot[i] = 0.0f; cnt[i] = 0.0f; }
    if (i == 0) { H[0] = 0x7f7fffffu; H[1] = 0u; }
}

// ---------------------------------------------------------------------------
// 1) per-edge L2 distance + global min/max (bit-pattern atomics: s > 0)
//    one wave per edge, coalesced 32-lane row reads
// ---------------------------------------------------------------------------
__global__ __launch_bounds__(256) void escore_kernel(
    const float* __restrict__ x, const int* __restrict__ src,
    const int* __restrict__ dst, float* __restrict__ s_raw, uint32_t* H) {
    int tid  = threadIdx.x;
    int lane = tid & 31;
    int wave = tid >> 5;
    int e    = blockIdx.x * 8 + wave;
    __shared__ uint32_t smin, smax;
    if (tid == 0) { smin = 0x7f7fffffu; smax = 0u; }
    __syncthreads();
    if (e < EE) {
        const float* pu = x + (size_t)src[e] * DD;
        const float* pv = x + (size_t)dst[e] * DD;
        __builtin_prefetch(pu + lane, 0, 1);   // global_prefetch_b8
        __builtin_prefetch(pv + lane, 0, 1);
        float acc = 0.0f;
#pragma unroll
        for (int k = 0; k < 8; ++k) {
            float d = pu[lane + 32 * k] - pv[lane + 32 * k];
            acc += d * d;
        }
#pragma unroll
        for (int off = 16; off > 0; off >>= 1) acc += __shfl_down(acc, off, 32);
        if (lane == 0) {
            float s = sqrtf(acc + 1e-12f);
            s_raw[e] = s;
            atomicMin(&smin, __float_as_uint(s));
            atomicMax(&smax, __float_as_uint(s));
        }
    }
    __syncthreads();
    if (tid == 0) { atomicMin(&H[0], smin); atomicMax(&H[1], smax); }
}

// ---------------------------------------------------------------------------
// 2) min-max normalize + per-node sums + packed edge records for grow loop
//    record: {x = src | dst<<16, y = bits(score)} -> one b64 load per edge
// ---------------------------------------------------------------------------
__global__ void norm_kernel(const float* __restrict__ s_raw, float* __restrict__ s_norm,
                            const int* __restrict__ src, const int* __restrict__ dst,
                            const uint32_t* H, float* tot, float* cnt,
                            uint2* __restrict__ ep) {
    int e = blockIdx.x * 256 + threadIdx.x;
    if (e >= EE) return;
    float mn = __uint_as_float(H[0]);
    float mx = __uint_as_float(H[1]);
    float s  = (s_raw[e] - mn) / (mx - mn) + SADD;
    s_norm[e] = s;
    int u = src[e], v = dst[e];
    ep[e] = make_uint2((uint32_t)u | ((uint32_t)v << 16), __float_as_uint(s));
    atomicAdd(&tot[u], s);  atomicAdd(&tot[v], s);
    atomicAdd(&cnt[u], 1.f); atomicAdd(&cnt[v], 1.f);
}

__global__ void avg_kernel(const float* tot, const float* cnt, float* avg) {
    int i = blockIdx.x * 256 + threadIdx.x;
    if (i < NN) avg[i] = tot[i] / cnt[i];
}

// ---------------------------------------------------------------------------
// 3) cluster growth: one block per seed, LDS-resident 4096-bit mask,
//    block argmin with lowest-edge-index tie-break (matches jnp.argmin)
// ---------------------------------------------------------------------------
__global__ __launch_bounds__(256) void grow_kernel(
    const uint2* __restrict__ ep, uint32_t* __restrict__ masks) {
    __shared__ uint32_t mask[MW];
    __shared__ float rs[256];
    __shared__ int   ri[256];
    __shared__ float s_total;
    __shared__ int   s_done;
    int tid  = threadIdx.x;
    int seed = blockIdx.x;
    if (tid < MW) mask[tid] = 0u;
    if (tid == 0) { s_total = 0.0f; s_done = 0; }
    __syncthreads();
    if (tid == 0) mask[seed >> 5] = (1u << (seed & 31));
    __syncthreads();

    for (;;) {
        float bs = __builtin_inff();
        int   bi = -1;
        for (int e = tid; e < EE; e += 256) {
            uint2 p = ep[e];
            int u = (int)(p.x & 0xffffu);
            int v = (int)(p.x >> 16);
            bool c = ((mask[u >> 5] >> (u & 31)) & 1u) &&
                     !((mask[v >> 5] >> (v & 31)) & 1u);
            if (c) {
                float s = __uint_as_float(p.y);
                if (s < bs) { bs = s; bi = e; }   // first-min kept (e ascending)
            }
        }
        rs[tid] = bs; ri[tid] = bi;
        __syncthreads();
        for (int off = 128; off > 0; off >>= 1) {
            if (tid < off) {
                float so = rs[tid + off]; int io = ri[tid + off];
                bool take = (io >= 0) &&
                            (ri[tid] < 0 || so < rs[tid] ||
                             (so == rs[tid] && io < ri[tid]));
                if (take) { rs[tid] = so; ri[tid] = io; }
            }
            __syncthreads();
        }
        if (tid == 0) {
            if (ri[0] < 0 || !(s_total < UB)) s_done = 1;
            else {
                int e = ri[0];
                int v = (int)(ep[e].x >> 16);
                mask[v >> 5] |= (1u << (v & 31));
                s_total += rs[0];
            }
        }
        __syncthreads();
        if (s_done) break;
    }
    for (int w = tid; w < MW; w += 256) masks[(size_t)seed * MW + w] = mask[w];
}

// ---------------------------------------------------------------------------
// 4) dedup: FNV hash per row, then first-occurrence uniqueness
// ---------------------------------------------------------------------------
__global__ void hash_kernel(const uint32_t* masks, unsigned long long* hash) {
    int i = blockIdx.x * 256 + threadIdx.x;
    if (i >= NN) return;
    const uint32_t* row = masks + (size_t)i * MW;
    unsigned long long h = 1469598103934665603ull;
    for (int w = 0; w < MW; ++w) { h ^= row[w]; h *= 1099511628211ull; }
    hash[i] = h;
}

__global__ void uniq_kernel(const uint32_t* masks, const unsigned long long* hash,
                            uint32_t* flag) {
    int i = blockIdx.x * 256 + threadIdx.x;
    if (i >= NN) return;
    unsigned long long hi = hash[i];
    const uint32_t* ri = masks + (size_t)i * MW;
    uint32_t f = 1;
    for (int j = 0; j < i && f; ++j) {
        if (hash[j] == hi) {
            const uint32_t* rj = masks + (size_t)j * MW;
            bool eq = true;
            for (int w = 0; w < MW && eq; ++w) eq = (ri[w] == rj[w]);
            if (eq) f = 0;
        }
    }
    flag[i] = f;
}

__global__ void compact_kernel(const uint32_t* flag, uint32_t* uniqlist, uint32_t* H) {
    if (threadIdx.x == 0 && blockIdx.x == 0) {
        uint32_t U = 0;
        for (int i = 0; i < NN; ++i) if (flag[i]) uniqlist[U++] = (uint32_t)i;
        H[2] = U;
    }
}

// ---------------------------------------------------------------------------
// 5) greedy max-uncovered cover (single block; popcount counts over bitmasks)
// ---------------------------------------------------------------------------
__global__ __launch_bounds__(1024) void greedy_kernel(
    const uint32_t* __restrict__ masks, const uint32_t* __restrict__ uniqlist,
    uint32_t* H, uint32_t* sel, uint32_t* alive) {
    __shared__ uint32_t covered[MW];
    __shared__ int bc[1024];
    __shared__ int br[1024];
    __shared__ int s_K, s_cov, s_pick;
    int tid = threadIdx.x;
    int U   = (int)H[2];
    for (int w = tid; w < MW; w += 1024) covered[w] = 0u;
    for (int r = tid; r < U; r += 1024) alive[r] = 1u;
    if (tid == 0) { s_K = 0; s_cov = 0; }
    __syncthreads();

    while (s_cov < NN) {
        int bestc = -1, bestr = 0x7fffffff;
        for (int r = tid; r < U; r += 1024) {
            if (!alive[r]) continue;
            const uint32_t* row = masks + (size_t)uniqlist[r] * MW;
            int cnt = 0;
            for (int w = 0; w < MW; ++w) cnt += __popc(row[w] & ~covered[w]);
            if (cnt > bestc || (cnt == bestc && r < bestr)) { bestc = cnt; bestr = r; }
        }
        bc[tid] = bestc; br[tid] = bestr;
        __syncthreads();
        for (int off = 512; off > 0; off >>= 1) {
            if (tid < off) {
                if (bc[tid + off] > bc[tid] ||
                    (bc[tid + off] == bc[tid] && br[tid + off] < br[tid])) {
                    bc[tid] = bc[tid + off]; br[tid] = br[tid + off];
                }
            }
            __syncthreads();
        }
        if (tid == 0) {
            s_pick = br[0];
            if (bc[0] <= 0) s_cov = NN;                 // no progress possible
            else { sel[s_K] = uniqlist[s_pick]; alive[s_pick] = 0u; ++s_K; }
        }
        __syncthreads();
        if (s_cov >= NN) break;
        {
            const uint32_t* row = masks + (size_t)uniqlist[s_pick] * MW;
            for (int w = tid; w < MW; w += 1024) covered[w] |= row[w];
        }
        __syncthreads();
        if (tid == 0) {
            int cc = 0;
            for (int w = 0; w < MW; ++w) cc += __popc(covered[w]);
            s_cov = cc;
        }
        __syncthreads();
    }
    if (tid == 0) H[3] = (uint32_t)s_K;
}

// ---------------------------------------------------------------------------
// 6) B = S @ adj as bitsets: B[b][v] = OR over edges (u,v) with u in mask_b
// ---------------------------------------------------------------------------
__global__ __launch_bounds__(256) void sadj_kernel(
    const uint2* __restrict__ ep,
    const uint32_t* __restrict__ masks, const uint32_t* __restrict__ sel,
    const uint32_t* H, uint32_t* __restrict__ Bbits) {
    int b = blockIdx.x;
    int K = (int)H[3];
    if (b >= K) return;
    __shared__ uint32_t m[MW];
    __shared__ uint32_t o[MW];
    int tid = threadIdx.x;
    const uint32_t* row = masks + (size_t)sel[b] * MW;
    for (int w = tid; w < MW; w += 256) { m[w] = row[w]; o[w] = 0u; }
    __syncthreads();
    for (int e = tid; e < EE; e += 256) {
        uint32_t uv = ep[e].x;
        int u = (int)(uv & 0xffffu);
        if ((m[u >> 5] >> (u & 31)) & 1u) {
            int v = (int)(uv >> 16);
            atomicOr(&o[v >> 5], 1u << (v & 31));
        }
    }
    __syncthreads();
    for (int w = tid; w < MW; w += 256) Bbits[(size_t)b * MW + w] = o[w];
}

// ---------------------------------------------------------------------------
// 7) adj_new[i][j] = any(B_i & mask_j), j != i; per-row nnz counts
// ---------------------------------------------------------------------------
__global__ __launch_bounds__(256) void adjnew_kernel(
    const uint32_t* __restrict__ Bbits, const uint32_t* __restrict__ masks,
    const uint32_t* __restrict__ sel, const uint32_t* H,
    uint32_t* __restrict__ adjbits, uint32_t* __restrict__ rowcnt) {
    int i = blockIdx.x;
    int K = (int)H[3];
    if (i >= K) return;
    __shared__ uint32_t Bi[MW];
    __shared__ uint32_t out[MW];
    int tid = threadIdx.x;
    for (int w = tid; w < MW; w += 256) { Bi[w] = Bbits[(size_t)i * MW + w]; out[w] = 0u; }
    __syncthreads();
    for (int j = tid; j < K; j += 256) {
        if (j == i) continue;                       // remove self loops
        const uint32_t* mj = masks + (size_t)sel[j] * MW;
        uint32_t any = 0u;
        for (int w = 0; w < MW; ++w) any |= (Bi[w] & mj[w]);
        if (any) atomicOr(&out[j >> 5], 1u << (j & 31));
    }
    __syncthreads();
    for (int w = tid; w < MW; w += 256) adjbits[(size_t)i * MW + w] = out[w];
    __syncthreads();
    if (tid == 0) {
        int c = 0;
        for (int w = 0; w < MW; ++w) c += __popc(out[w]);
        rowcnt[i] = (uint32_t)c;
    }
}

// ---------------------------------------------------------------------------
// 8) row-offset scan -> Enew, output offsets
// ---------------------------------------------------------------------------
__global__ void scan_kernel(uint32_t* H, const uint32_t* rowcnt, uint32_t* rowoff) {
    if (threadIdx.x == 0 && blockIdx.x == 0) {
        int K = (int)H[3];
        uint32_t acc = 0;
        for (int i = 0; i < K; ++i) { rowoff[i] = acc; acc += rowcnt[i]; }
        uint32_t Enew = acc;
        H[4] = Enew;
        uint32_t o1 = 2u * Enew;
        uint32_t o2 = o1 + (uint32_t)K * DD;
        uint32_t o3 = o2 + (uint32_t)K * NN;
        uint32_t o4 = o3 + (uint32_t)K * NN;
        H[5] = o1; H[6] = o2; H[7] = o3; H[8] = o4;
    }
}

// ---------------------------------------------------------------------------
// 9) emit edge_index_new (row-major nonzero order) as floats
// ---------------------------------------------------------------------------
__global__ __launch_bounds__(256) void wedges_kernel(
    const uint32_t* __restrict__ adjbits, const uint32_t* __restrict__ rowoff,
    const uint32_t* H, float* __restrict__ out, int out_size) {
    int i = blockIdx.x;
    int K = (int)H[3];
    if (i >= K) return;
    long Enew = (long)H[4];
    const uint32_t* row = adjbits + (size_t)i * MW;
    for (int j = threadIdx.x; j < K; j += 256) {
        if ((row[j >> 5] >> (j & 31)) & 1u) {
            int rank = 0, wj = j >> 5;
            for (int w = 0; w < wj; ++w) rank += __popc(row[w]);
            rank += __popc(row[wj] & ((1u << (j & 31)) - 1u));
            long pos = (long)rowoff[i] + rank;
            if (pos >= 0 && pos < out_size) out[pos] = (float)i;
            long pos2 = Enew + pos;
            if (pos2 >= 0 && pos2 < out_size) out[pos2] = (float)j;
        }
    }
}

// ---------------------------------------------------------------------------
// 10) emit S and Ms = S * avg
// ---------------------------------------------------------------------------
__global__ __launch_bounds__(256) void wSMs_kernel(
    const uint32_t* __restrict__ masks, const uint32_t* __restrict__ sel,
    const float* __restrict__ avg, const uint32_t* H,
    float* __restrict__ out, int out_size) {
    int i = blockIdx.x;
    int K = (int)H[3];
    if (i >= K) return;
    long o2 = (long)H[6], o3 = (long)H[7];
    const uint32_t* row = masks + (size_t)sel[i] * MW;
    for (int n = threadIdx.x; n < NN; n += 256) {
        float b = (float)((row[n >> 5] >> (n & 31)) & 1u);
        long p1 = o2 + (long)i * NN + n;
        long p2 = o3 + (long)i * NN + n;
        if (p1 >= 0 && p1 < out_size) out[p1] = b;
        if (p2 >= 0 && p2 < out_size) out[p2] = b * avg[n];
    }
}

// ---------------------------------------------------------------------------
// 11) x_new = Ms @ x via v_wmma_f32_16x16x32_f16, one wave per 16x16 tile.
//     A tiles (Ms) synthesized BRANCH-FREE: one mask word covers the whole
//     32-wide K-window (n0 is 32-aligned); avg fetched as four aligned
//     float4 runs; element = avg * (0/1 bit). No EXEC-masked loads.
// ---------------------------------------------------------------------------
__global__ __launch_bounds__(32) void xnew_wmma_kernel(
    const float* __restrict__ x, const float* __restrict__ avg,
    const uint32_t* __restrict__ masks, const uint32_t* __restrict__ sel,
    const uint32_t* H, float* __restrict__ out, int out_size) {
    const int K  = (int)H[3];
    const int i0 = blockIdx.y * 16;
    if (i0 >= K) return;                       // uniform exit, EXEC stays all-1s
    const int d0      = blockIdx.x * 16;
    const int lane    = threadIdx.x;
    const int halfsel = lane >> 4;             // 0: lanes 0-15, 1: lanes 16-31
    const int lm      = lane & 15;

    const int  irow     = i0 + lm;
    const bool rowValid = irow < K;
    const uint32_t srow = rowValid ? sel[irow] : 0u;
    const uint32_t* mrow = masks + (size_t)srow * MW;

    const int kbaseA = halfsel * 8;            // A: K in {kb..kb+7} U {kb+16..kb+23}
    const int kbaseB = halfsel * 8;            // B: K in {kb..kb+7} U {kb+16..kb+23}

    v8f c = {};
    for (int n0 = 0; n0 < NN; n0 += 32) {
        // one 32-bit mask word covers bits n0..n0+31; zero it for invalid rows
        uint32_t mword = mrow[n0 >> 5];
        if (!rowValid) mword = 0u;

        // avg runs: [n0+kbaseA, +8) and [n0+kbaseA+16, +8)  (32B-aligned)
        const float4* a0 = (const float4*)(avg + n0 + kbaseA);
        const float4* a1 = (const float4*)(avg + n0 + kbaseA + 16);
        float4 r0 = a0[0], r1 = a0[1];
        float4 r2 = a1[0], r3 = a1[1];
        float avv[16] = { r0.x, r0.y, r0.z, r0.w, r1.x, r1.y, r1.z, r1.w,
                          r2.x, r2.y, r2.z, r2.w, r3.x, r3.y, r3.z, r3.w };

        v16h a, b;
#pragma unroll
        for (int h = 0; h < 16; ++h) {
            int kk = kbaseA + ((h < 8) ? h : (h + 8));       // bit index in mword
            float bit = (float)((mword >> kk) & 1u);
            a[h] = (half_t)(avv[h] * bit);                   // Ms[i0+lm][n0+kk]
        }
#pragma unroll
        for (int h = 0; h < 16; ++h) {
            int kk = kbaseB + ((h < 8) ? h : (h + 8));
            b[h] = (half_t)x[(size_t)(n0 + kk) * DD + (d0 + lm)];  // x[n][d0+lm]
        }
        // (neg_a, A, neg_b, B, c_mod, C, reuse_a, reuse_b)
        c = __builtin_amdgcn_wmma_f32_16x16x32_f16(
                false, a, false, b, (short)0, c, false, false);
    }

    const long o1 = (long)H[5];
#pragma unroll
    for (int r = 0; r < 8; ++r) {
        int m   = r + halfsel * 8;             // C/D: VGPR r -> M=r (lo) / M=8+r (hi)
        int row = i0 + m;
        if (row < K) {
            long idx = o1 + (long)row * DD + (d0 + lm);
            if (idx >= 0 && idx < out_size) out[idx] = c[r];
        }
    }
}

// ---------------------------------------------------------------------------
// 12) copy original edge_index (as floats) to the tail of d_out
// ---------------------------------------------------------------------------
__global__ void wei_kernel(const int* __restrict__ ei, const uint32_t* H,
                           float* __restrict__ out, int out_size) {
    int t = blockIdx.x * 256 + threadIdx.x;
    if (t >= 2 * EE) return;
    long p = (long)H[8] + t;
    if (p >= 0 && p < out_size) out[p] = (float)ei[t];
}

// ---------------------------------------------------------------------------
// Host-side launch (graph-capture safe: only kernel launches on `stream`)
// ---------------------------------------------------------------------------
extern "C" void kernel_launch(void* const* d_in, const int* in_sizes, int n_in,
                              void* d_out, int out_size, void* d_ws, size_t ws_size,
                              hipStream_t stream) {
    (void)in_sizes; (void)n_in; (void)ws_size;
    const float* x  = (const float*)d_in[0];
    const int*   ei = (const int*)d_in[1];
    const int*   src = ei;
    const int*   dst = ei + EE;
    float* out = (float*)d_out;

    // workspace carve-out (256 B aligned slices)
    uint8_t* W = (uint8_t*)d_ws;
    size_t off = 0;
    auto carve = [&](size_t bytes) -> void* {
        void* p = (void*)(W + off);
        off = (off + bytes + 255) & ~(size_t)255;
        return p;
    };
    uint32_t*           H        = (uint32_t*)carve(64 * 4);
    float*              s_raw    = (float*)carve((size_t)EE * 4);
    float*              s_norm   = (float*)carve((size_t)EE * 4);
    uint2*              ep       = (uint2*)carve((size_t)EE * 8);
    float*              tot      = (float*)carve((size_t)NN * 4);
    float*              cnt      = (float*)carve((size_t)NN * 4);
    float*              avg      = (float*)carve((size_t)NN * 4);
    uint32_t*           masks    = (uint32_t*)carve((size_t)NN * MW * 4);
    unsigned long long* hash     = (unsigned long long*)carve((size_t)NN * 8);
    uint32_t*           uflag    = (uint32_t*)carve((size_t)NN * 4);
    uint32_t*           uniqlist = (uint32_t*)carve((size_t)NN * 4);
    uint32_t*           alive    = (uint32_t*)carve((size_t)NN * 4);
    uint32_t*           sel      = (uint32_t*)carve((size_t)NN * 4);
    uint32_t*           Bbits    = (uint32_t*)carve((size_t)NN * MW * 4);
    uint32_t*           adjbits  = (uint32_t*)carve((size_t)NN * MW * 4);
    uint32_t*           rowcnt   = (uint32_t*)carve((size_t)NN * 4);
    uint32_t*           rowoff   = (uint32_t*)carve((size_t)(NN + 1) * 4);

    init_kernel<<<NN / 256, 256, 0, stream>>>(H, tot, cnt);
    escore_kernel<<<EE / 8, 256, 0, stream>>>(x, src, dst, s_raw, H);
    norm_kernel<<<EE / 256, 256, 0, stream>>>(s_raw, s_norm, src, dst, H, tot, cnt, ep);
    avg_kernel<<<NN / 256, 256, 0, stream>>>(tot, cnt, avg);
    grow_kernel<<<NN, 256, 0, stream>>>(ep, masks);
    hash_kernel<<<NN / 256, 256, 0, stream>>>(masks, hash);
    uniq_kernel<<<NN / 256, 256, 0, stream>>>(masks, hash, uflag);
    compact_kernel<<<1, 32, 0, stream>>>(uflag, uniqlist, H);
    greedy_kernel<<<1, 1024, 0, stream>>>(masks, uniqlist, H, sel, alive);
    sadj_kernel<<<NN, 256, 0, stream>>>(ep, masks, sel, H, Bbits);
    adjnew_kernel<<<NN, 256, 0, stream>>>(Bbits, masks, sel, H, adjbits, rowcnt);
    scan_kernel<<<1, 32, 0, stream>>>(H, rowcnt, rowoff);
    wedges_kernel<<<NN, 256, 0, stream>>>(adjbits, rowoff, H, out, out_size);
    wSMs_kernel<<<NN, 256, 0, stream>>>(masks, sel, avg, H, out, out_size);
    {
        dim3 grid(DD / 16, NN / 16);  // col tiles x row tiles (rows guarded by K)
        xnew_wmma_kernel<<<grid, 32, 0, stream>>>(x, avg, masks, sel, H, out, out_size);
    }
    wei_kernel<<<(2 * EE) / 256, 256, 0, stream>>>(ei, H, out, out_size);
}
